// MMDLoss_71829033058812
// MI455X (gfx1250) — matrix operand order repaired and checked by reading, not compiled, not run on hardware
//
#include <hip/hip_runtime.h>

// ---------------------------------------------------------------------------
// MMD loss for MI455X (gfx1250, wave32).
// G = Z Z^T with Z = [X;Y] (512 x 100352), upper-triangular 64x64 blocks only.
// Preferred path: pre-convert Z to bf16 in d_ws (98 MB, L2-resident on the
// 192 MB L2), then a pure-WMMA GEMM (v_wmma_f32_16x16x32_bf16, f32 accum,
// K-split + global_atomic_add_f32). Fallback path converts on the fly.
// ---------------------------------------------------------------------------

typedef __attribute__((ext_vector_type(16))) __bf16 v16bf;
typedef __attribute__((ext_vector_type(8)))  __bf16 v8bf;
typedef __attribute__((ext_vector_type(8)))  float  v8f;
typedef __attribute__((ext_vector_type(4)))  float  v4f;   // native vector for nontemporal builtins

#define D_DIM   100352      // 512*14*14
#define NROWS   512
#define KSPLIT  14
#define KCHUNK  7168        // 100352 / 14
#define NPOS    36          // upper-triangular 8x8 grid of 64x64 blocks

__device__ __forceinline__ const float* zrow(const float* x, const float* y, int r) {
    return (r < 256) ? (x + (size_t)r * D_DIM) : (y + (size_t)(r - 256) * D_DIM);
}

__device__ __forceinline__ v8f wmma_bf16(v16bf a, v16bf b, v8f c) {
    return __builtin_amdgcn_wmma_f32_16x16x32_bf16(
        /*neg_a=*/false, a, /*neg_b=*/false, b,
        /*c_mod=*/(short)0, c, /*reuse_a=*/false, /*reuse_b=*/false);
}

__device__ __forceinline__ void atom_add(float* p, float v) {
    (void)__hip_atomic_fetch_add(p, v, __ATOMIC_RELAXED, __HIP_MEMORY_SCOPE_AGENT);
}

__global__ void zero_gram(float* __restrict__ g) {
    int i = blockIdx.x * blockDim.x + threadIdx.x;
    if (i < NROWS * NROWS) g[i] = 0.0f;
}

// --------------------------- pre-conversion pass ---------------------------
// Z[row][k] (bf16) from X/Y (f32). Non-temporal input loads: the f32 stream is
// read once and must not evict the L2-resident bf16 output.
__global__ __launch_bounds__(256) void convert_bf16(const float* __restrict__ x,
                                                    const float* __restrict__ y,
                                                    __bf16* __restrict__ z) {
    int row = blockIdx.y;
    int k   = (blockIdx.x * 256 + threadIdx.x) * 8;   // 49*256*8 == 100352
    const float* src = zrow(x, y, row) + k;
    const v4f* s4 = (const v4f*)src;
    v4f a = __builtin_nontemporal_load(s4);
    v4f b = __builtin_nontemporal_load(s4 + 1);
    v8bf r;
    r[0] = (__bf16)a[0]; r[1] = (__bf16)a[1]; r[2] = (__bf16)a[2]; r[3] = (__bf16)a[3];
    r[4] = (__bf16)b[0]; r[5] = (__bf16)b[1]; r[6] = (__bf16)b[2]; r[7] = (__bf16)b[3];
    *(v8bf*)(z + (size_t)row * D_DIM + k) = r;
}

// ----------------------------- bf16-direct GEMM ----------------------------
// A fragment, 16x32 bf16: lane (h*16+t) holds row t: elems 0..15 = K c..c+15
// contiguous, c = k0 + 16*h. Two 16B chunks.
__device__ __forceinline__ v16bf load_af(const __bf16* row, int c) {
    v8bf lo = *(const v8bf*)(row + c);
    v8bf hi = *(const v8bf*)(row + c + 8);
    return __builtin_shufflevector(lo, hi, 0, 1, 2, 3, 4, 5, 6, 7,
                                           8, 9, 10, 11, 12, 13, 14, 15);
}
// B fragment, 32x16 bf16: lane (h*16+n) holds column n, K = c..c+15
// contiguous, c = k0 + 16*h. One 32B chunk.
__device__ __forceinline__ v16bf load_bf(const __bf16* row, int c) {
    return *(const v16bf*)(row + c);
}

__global__ __launch_bounds__(128) void mmd_gram_bf16(const __bf16* __restrict__ z,
                                                     float* __restrict__ gram) {
    int wg = blockIdx.x;
    int s  = wg % KSPLIT;
    int p  = wg / KSPLIT;
    int bi = 0, rem = p, rowlen = 8;
    while (rem >= rowlen) { rem -= rowlen; rowlen--; bi++; }
    int bj = bi + rem;

    int tid  = threadIdx.x;
    int wave = tid >> 5;
    int lane = tid & 31;
    int wm   = wave >> 1;
    int wn   = wave & 1;
    int h    = lane >> 4;
    int t    = lane & 15;

    int rowA = bi * 64 + wm * 32;
    int rowB = bj * 64 + wn * 32;

    const __bf16* ar0 = z + (size_t)(rowA + t) * D_DIM;
    const __bf16* ar1 = z + (size_t)(rowA + 16 + t) * D_DIM;
    const __bf16* br0 = z + (size_t)(rowB + t) * D_DIM;
    const __bf16* br1 = z + (size_t)(rowB + 16 + t) * D_DIM;

    int ca = 16 * h;
    int cb = 16 * h;

    v8f acc00 = {}, acc01 = {}, acc10 = {}, acc11 = {};

    int k0   = s * KCHUNK;
    int kend = k0 + KCHUNK;
#pragma unroll 2
    for (; k0 < kend; k0 += 32) {
        v16bf a0 = load_af(ar0, k0 + ca);
        v16bf a1 = load_af(ar1, k0 + ca);
        v16bf b0 = load_bf(br0, k0 + cb);
        v16bf b1 = load_bf(br1, k0 + cb);
        acc00 = wmma_bf16(a0, b0, acc00);
        acc01 = wmma_bf16(a0, b1, acc01);
        acc10 = wmma_bf16(a1, b0, acc10);
        acc11 = wmma_bf16(a1, b1, acc11);
    }

    int rbase = rowA + 8 * h;
    int cbase = rowB + t;
#pragma unroll
    for (int v = 0; v < 8; v++) {
        atom_add(gram + (size_t)(rbase      + v) * NROWS + cbase,      acc00[v]);
        atom_add(gram + (size_t)(rbase      + v) * NROWS + cbase + 16, acc01[v]);
        atom_add(gram + (size_t)(rbase + 16 + v) * NROWS + cbase,      acc10[v]);
        atom_add(gram + (size_t)(rbase + 16 + v) * NROWS + cbase + 16, acc11[v]);
    }
}

// ------------------- fallback GEMM (on-the-fly f32->bf16) ------------------
__device__ __forceinline__ v16bf cvt16(v4f q0, v4f q1, v4f q2, v4f q3) {
    v16bf r;
    r[0]  = (__bf16)q0[0]; r[1]  = (__bf16)q0[1]; r[2]  = (__bf16)q0[2]; r[3]  = (__bf16)q0[3];
    r[4]  = (__bf16)q1[0]; r[5]  = (__bf16)q1[1]; r[6]  = (__bf16)q1[2]; r[7]  = (__bf16)q1[3];
    r[8]  = (__bf16)q2[0]; r[9]  = (__bf16)q2[1]; r[10] = (__bf16)q2[2]; r[11] = (__bf16)q2[3];
    r[12] = (__bf16)q3[0]; r[13] = (__bf16)q3[1]; r[14] = (__bf16)q3[2]; r[15] = (__bf16)q3[3];
    return r;
}
__device__ __forceinline__ v16bf load_afrag(const float* row, int c) {
    const v4f* p0 = (const v4f*)(row + c);
    const v4f* p1 = (const v4f*)(row + c + 8);
    return cvt16(p0[0], p0[1], p1[0], p1[1]);
}
__device__ __forceinline__ v16bf load_bfrag(const float* row, int c) {
    const v4f* p = (const v4f*)(row + c);
    return cvt16(p[0], p[1], p[2], p[3]);
}

__global__ __launch_bounds__(128) void mmd_gram_f32(const float* __restrict__ x,
                                                    const float* __restrict__ y,
                                                    float* __restrict__ gram) {
    int wg = blockIdx.x;
    int s  = wg % KSPLIT;
    int p  = wg / KSPLIT;
    int bi = 0, rem = p, rowlen = 8;
    while (rem >= rowlen) { rem -= rowlen; rowlen--; bi++; }
    int bj = bi + rem;

    int tid  = threadIdx.x;
    int wave = tid >> 5;
    int lane = tid & 31;
    int wm   = wave >> 1;
    int wn   = wave & 1;
    int h    = lane >> 4;
    int t    = lane & 15;

    int rowA = bi * 64 + wm * 32;
    int rowB = bj * 64 + wn * 32;

    const float* ar0 = zrow(x, y, rowA + t);
    const float* ar1 = zrow(x, y, rowA + 16 + t);
    const float* br0 = zrow(x, y, rowB + t);
    const float* br1 = zrow(x, y, rowB + 16 + t);

    int ca = 16 * h;
    int cb = 16 * h;

    v8f acc00 = {}, acc01 = {}, acc10 = {}, acc11 = {};

    int k0   = s * KCHUNK;
    int kend = k0 + KCHUNK;
    for (; k0 < kend; k0 += 32) {
        v16bf a0 = load_afrag(ar0, k0 + ca);
        v16bf a1 = load_afrag(ar1, k0 + ca);
        v16bf b0 = load_bfrag(br0, k0 + cb);
        v16bf b1 = load_bfrag(br1, k0 + cb);
        acc00 = wmma_bf16(a0, b0, acc00);
        acc01 = wmma_bf16(a0, b1, acc01);
        acc10 = wmma_bf16(a1, b0, acc10);
        acc11 = wmma_bf16(a1, b1, acc11);
    }

    int rbase = rowA + 8 * h;
    int cbase = rowB + t;
#pragma unroll
    for (int v = 0; v < 8; v++) {
        atom_add(gram + (size_t)(rbase      + v) * NROWS + cbase,      acc00[v]);
        atom_add(gram + (size_t)(rbase      + v) * NROWS + cbase + 16, acc01[v]);
        atom_add(gram + (size_t)(rbase + 16 + v) * NROWS + cbase,      acc10[v]);
        atom_add(gram + (size_t)(rbase + 16 + v) * NROWS + cbase + 16, acc11[v]);
    }
}

// -------------------------------- reduction --------------------------------
__global__ __launch_bounds__(1024) void mmd_reduce(const float* __restrict__ g,
                                                   float* __restrict__ out) {
    __shared__ float diag[NROWS];
    __shared__ float red[1024];
    int tid = threadIdx.x;
    for (int i = tid; i < NROWS; i += 1024) diag[i] = g[(size_t)i * NROWS + i];
    __syncthreads();

    const float c = 1.0f / (2.0f * (float)D_DIM);  // (d2/D) / (2*sigma^2), sigma=1
    float sum = 0.0f;
    for (int idx = tid; idx < 256 * 256; idx += 1024) {
        int i = idx >> 8;
        int j = idx & 255;
        int lo = min(i, j), hi = max(i, j);
        float gxx = g[(size_t)lo * NROWS + hi];
        float gyy = g[(size_t)(256 + lo) * NROWS + 256 + hi];
        float gxy = g[(size_t)i * NROWS + 256 + j];    // always upper triangle
        float d2xx = fmaxf(diag[i]       + diag[j]       - 2.0f * gxx, 0.0f);
        float d2yy = fmaxf(diag[256 + i] + diag[256 + j] - 2.0f * gyy, 0.0f);
        float d2xy = fmaxf(diag[i]       + diag[256 + j] - 2.0f * gxy, 0.0f);
        sum += __expf(-d2xx * c) + __expf(-d2yy * c) - 2.0f * __expf(-d2xy * c);
    }
    red[tid] = sum;
    __syncthreads();
    for (int off = 512; off > 0; off >>= 1) {
        if (tid < off) red[tid] += red[tid + off];
        __syncthreads();
    }
    if (tid == 0) out[0] = red[0] * (1.0f / 65536.0f);
}

extern "C" void kernel_launch(void* const* d_in, const int* in_sizes, int n_in,
                              void* d_out, int out_size, void* d_ws, size_t ws_size,
                              hipStream_t stream) {
    (void)in_sizes; (void)n_in; (void)out_size;
    const float* x = (const float*)d_in[0];
    const float* y = (const float*)d_in[1];
    float* gram = (float*)d_ws;                          // 1 MB
    float* out  = (float*)d_out;

    const size_t gram_bytes = (size_t)NROWS * NROWS * sizeof(float);
    const size_t z_bytes    = (size_t)NROWS * D_DIM * sizeof(__bf16);

    zero_gram<<<(NROWS * NROWS + 255) / 256, 256, 0, stream>>>(gram);

    if (ws_size >= gram_bytes + z_bytes) {
        __bf16* z = (__bf16*)((char*)d_ws + gram_bytes);
        convert_bf16<<<dim3(49, NROWS), 256, 0, stream>>>(x, y, z);
        mmd_gram_bf16<<<NPOS * KSPLIT, 128, 0, stream>>>(z, gram);
    } else {
        mmd_gram_f32<<<NPOS * KSPLIT, 128, 0, stream>>>(x, y, gram);
    }
    mmd_reduce<<<1, 1024, 0, stream>>>(gram, out);
}
